// MultiHeadAttention_4544075399598
// MI455X (gfx1250) — matrix operand order repaired and checked by reading, not compiled, not run on hardware
//
#include <hip/hip_runtime.h>
#include <hip/hip_bf16.h>

// MultiHeadAttention forward for MI455X (gfx1250, wave32, WMMA).
// B=4, S=2048, D=2048, H=16, HD=128. Compute-bound => everything on
// v_wmma_f32_16x16x32_f16 with f32 accumulation.

#define B_DIM  4
#define S_DIM  2048
#define D_DIM  2048
#define H_NUM  16
#define HD_DIM 128

typedef __attribute__((ext_vector_type(16))) _Float16 v16h;
typedef __attribute__((ext_vector_type(8)))  _Float16 v8h;
typedef __attribute__((ext_vector_type(8)))  float    v8f;

// ---------------------------------------------------------------------------
// WMMA wrapper: D = A(16x32 f16) * B(32x16 f16) + C(16x16 f32)
// ---------------------------------------------------------------------------
__device__ __forceinline__ v8f wmma_f16(v16h a, v16h b, v8f c) {
  return __builtin_amdgcn_wmma_f32_16x16x32_f16(
      /*neg_a=*/false, a, /*neg_b=*/false, b,
      /*c_mod=*/(short)0, c, /*reuse_a=*/false, /*reuse_b=*/false);
}

// A-fragment (16x32 f16) from row-major A, lda in elements.
// Lane layout: row = lane%16; lanes 0-15 hold K [0..7]+[16..23],
// lanes 16-31 hold K [8..15]+[24..31]  -> two contiguous 16B loads.
__device__ __forceinline__ v16h load_a_frag(const _Float16* __restrict__ A,
                                            int lda, int row, int kBase, int half_) {
  const int c0 = kBase + (half_ ? 8 : 0);
  const _Float16* p = A + (size_t)row * lda + c0;
  v8h lo = *(const v8h*)(p);
  v8h hi = *(const v8h*)(p + 16);
  v16h r;
#pragma unroll
  for (int i = 0; i < 8; ++i) { r[i] = lo[i]; r[i + 8] = hi[i]; }
  return r;
}

// ---------------------------------------------------------------------------
// f32 -> f16 conversion (grid-stride)
// ---------------------------------------------------------------------------
__global__ void cvt_f32_to_f16(const float* __restrict__ in,
                               _Float16* __restrict__ out, size_t n) {
  size_t i = (size_t)blockIdx.x * blockDim.x + threadIdx.x;
  size_t stride = (size_t)gridDim.x * blockDim.x;
  for (; i < n; i += stride) out[i] = (_Float16)in[i];
}

// ---------------------------------------------------------------------------
// Tiled WMMA GEMM: C[M,N] = A[M,K] @ B[K,N], A/B row-major f16.
// Workgroup = 8 waves (4 along M x 2 along N), 128x128 output tile.
// Each wave: 32x64 = 2x4 C fragments, K-loop in steps of 32, unrolled x2.
// MODE 0: write Q  (f16, scaled, [b,h,s,d])
// MODE 1: write Kt (f16, transposed, [b,h,d,s])
// MODE 2: write V  (f16, [b,h,s,d])
// MODE 3: write f32 out + bias ([M,N] row-major)
// ---------------------------------------------------------------------------
template <int MODE>
__global__ __launch_bounds__(256) void gemm128(const _Float16* __restrict__ A,
                                               const _Float16* __restrict__ Bm,
                                               _Float16* __restrict__ outH,
                                               float* __restrict__ outF,
                                               const float* __restrict__ bias,
                                               int M, int N, int K, float scale) {
  const int tid   = threadIdx.x;
  const int wave  = tid >> 5;
  const int lane  = tid & 31;
  const int half_ = lane >> 4;
  const int ncol  = lane & 15;
  const int waveM = wave & 3;
  const int waveN = wave >> 2;
  const int mBase = blockIdx.y * 128 + waveM * 32;
  const int nBase = blockIdx.x * 128 + waveN * 64;

  v8f acc[2][4];
#pragma unroll
  for (int i = 0; i < 2; ++i)
#pragma unroll
    for (int j = 0; j < 4; ++j) acc[i][j] = {};

#pragma unroll 2
  for (int kb = 0; kb < K; kb += 32) {
    v16h a[2], b[4];
#pragma unroll
    for (int i = 0; i < 2; ++i)
      a[i] = load_a_frag(A, K, mBase + i * 16 + ncol, kb, half_);
    // B fragment: column = lane%16, K in table order over elements.
#pragma unroll
    for (int j = 0; j < 4; ++j)
#pragma unroll
      for (int s = 0; s < 16; ++s)
        b[j][s] = Bm[(size_t)(kb + (s & 7) + 8 * half_ + ((s >> 3) << 4)) * N + nBase + j * 16 + ncol];

    if (kb + 32 < K) {  // gfx1250 global_prefetch for next K slab
      __builtin_prefetch(A + (size_t)(mBase + ncol) * K + kb + 32, 0, 1);
      __builtin_prefetch(Bm + (size_t)(kb + 32 + lane) * N + nBase, 0, 1);
    }

#pragma unroll
    for (int i = 0; i < 2; ++i)
#pragma unroll
      for (int j = 0; j < 4; ++j)
        acc[i][j] = wmma_f16(a[i], b[j], acc[i][j]);
  }

  // Epilogue. C layout: VGPR r, half -> M = r + 8*half, N = ncol.
#pragma unroll
  for (int i = 0; i < 2; ++i) {
#pragma unroll
    for (int j = 0; j < 4; ++j) {
#pragma unroll
      for (int r = 0; r < 8; ++r) {
        float val = acc[i][j][r];
        int gr = mBase + i * 16 + r + 8 * half_;
        int gc = nBase + j * 16 + ncol;
        if (MODE == 3) {
          outF[(size_t)gr * N + gc] = val + bias[gc];
        } else {
          int bb = gr / S_DIM, s = gr % S_DIM;
          int h = gc >> 7, d = gc & 127;
          if (MODE == 0)
            outH[((((size_t)bb * H_NUM + h) * S_DIM + s) << 7) + d] =
                (_Float16)(val * scale);
          else if (MODE == 1)  // transposed: [b,h,d,s]
            outH[(((size_t)bb * H_NUM + h) * HD_DIM + d) * S_DIM + s] =
                (_Float16)val;
          else
            outH[((((size_t)bb * H_NUM + h) * S_DIM + s) << 7) + d] =
                (_Float16)val;
        }
      }
    }
  }
}

// ---------------------------------------------------------------------------
// Causal flash attention. Block = 4 waves; each wave owns 16 q-rows of one
// (b,h). 64 kv columns per online-softmax update (amortizes the VALU-side
// reductions/rescale against 32 WMMAs per iteration; XDL WMMA co-executes
// with VALU). P re-layout C->A via per-wave LDS tile.
// ---------------------------------------------------------------------------
__global__ __launch_bounds__(128) void attn_fa(const _Float16* __restrict__ q,
                                               const _Float16* __restrict__ kt,
                                               const _Float16* __restrict__ v,
                                               _Float16* __restrict__ ctx) {
  __shared__ _Float16 Pl[4][16][64];  // per-wave 16x64 P tile (8 KB total)

  const int tid   = threadIdx.x;
  const int wave  = tid >> 5;
  const int lane  = tid & 31;
  const int half_ = lane >> 4;
  const int ncol  = lane & 15;

  const int qBlocks = S_DIM / 64;
  const int bh = blockIdx.x / qBlocks;
  const int qb = blockIdx.x % qBlocks;
  const int q0 = qb * 64 + wave * 16;
  const int b  = bh >> 4;
  const int h  = bh & 15;

  const _Float16* qp  = q  + (size_t)bh * S_DIM * HD_DIM;  // [s,d]
  const _Float16* ktp = kt + (size_t)bh * HD_DIM * S_DIM;  // [d,s]
  const _Float16* vp  = v  + (size_t)bh * S_DIM * HD_DIM;  // [s,d]

  // Q fragments for this wave's 16 rows (4 d-chunks of 32). Q pre-scaled.
  v16h qf[4];
#pragma unroll
  for (int c = 0; c < 4; ++c)
    qf[c] = load_a_frag(qp, HD_DIM, q0 + ncol, c * 32, half_);

  float mstate[8], lstate[8];
#pragma unroll
  for (int r = 0; r < 8; ++r) { mstate[r] = -1e30f; lstate[r] = 0.f; }
  v8f cacc[8];
#pragma unroll
  for (int t = 0; t < 8; ++t) cacc[t] = {};

  for (int j = 0; j < q0 + 16; j += 64) {
    // ---- scores: four 16x16 tiles over kv cols [j .. j+63] ----
    v8f sc[4];
#pragma unroll
    for (int f = 0; f < 4; ++f) sc[f] = {};
#pragma unroll
    for (int c = 0; c < 4; ++c) {
      const _Float16* kr = ktp + (size_t)(c * 32) * S_DIM + j + ncol;
#pragma unroll
      for (int f = 0; f < 4; ++f) {
        v16h bf;
#pragma unroll
        for (int s = 0; s < 16; ++s)
          bf[s] = kr[(size_t)((s & 7) + 8 * half_ + ((s >> 3) << 4)) * S_DIM + 16 * f];
        sc[f] = wmma_f16(qf[c], bf, sc[f]);
      }
    }
    // causal mask
    if (j + 63 > q0) {
#pragma unroll
      for (int r = 0; r < 8; ++r) {
        int qrow = q0 + r + 8 * half_;
#pragma unroll
        for (int f = 0; f < 4; ++f)
          if (j + 16 * f + ncol > qrow) sc[f][r] = -1e30f;
      }
    }
    // ---- online softmax (row reductions across each 16-lane half) ----
#pragma unroll
    for (int r = 0; r < 8; ++r) {
      float x0 = sc[0][r], x1 = sc[1][r], x2 = sc[2][r], x3 = sc[3][r];
      float mx = fmaxf(fmaxf(x0, x1), fmaxf(x2, x3));
#pragma unroll
      for (int msk = 8; msk >= 1; msk >>= 1)
        mx = fmaxf(mx, __shfl_xor(mx, msk, 32));
      float mnew  = fmaxf(mstate[r], mx);
      float p0    = __expf(x0 - mnew);
      float p1    = __expf(x1 - mnew);
      float p2    = __expf(x2 - mnew);
      float p3    = __expf(x3 - mnew);
      float alpha = __expf(mstate[r] - mnew);
      mstate[r] = mnew;
      float sum = (p0 + p1) + (p2 + p3);
#pragma unroll
      for (int msk = 8; msk >= 1; msk >>= 1)
        sum += __shfl_xor(sum, msk, 32);
      lstate[r] = lstate[r] * alpha + sum;
#pragma unroll
      for (int t = 0; t < 8; ++t) cacc[t][r] = cacc[t][r] * alpha;
      // stash P in C-layout into LDS
      _Float16* pw = &Pl[wave][r + 8 * half_][ncol];
      pw[0]  = (_Float16)p0;
      pw[16] = (_Float16)p1;
      pw[32] = (_Float16)p2;
      pw[48] = (_Float16)p3;
    }
    // wave-private LDS store->load ordering
    asm volatile("s_wait_dscnt 0" ::: "memory");
    // read P back as two A-fragments (kv chunks [j..j+31], [j+32..j+63])
    v16h pa[2];
#pragma unroll
    for (int c2 = 0; c2 < 2; ++c2) {
      const _Float16* pr = &Pl[wave][ncol][c2 * 32 + (half_ ? 8 : 0)];
      v8h lo = *(const v8h*)(pr);
      v8h hi = *(const v8h*)(pr + 16);
#pragma unroll
      for (int i = 0; i < 8; ++i) { pa[c2][i] = lo[i]; pa[c2][i + 8] = hi[i]; }
    }
    asm volatile("" ::: "memory");
    // ---- ctx += P @ V : 8 d-tiles x 2 kv chunks ----
#pragma unroll
    for (int t = 0; t < 8; ++t) {
#pragma unroll
      for (int c2 = 0; c2 < 2; ++c2) {
        v16h vb;
#pragma unroll
        for (int s = 0; s < 16; ++s)
          vb[s] = vp[(size_t)(j + 32 * c2 + (s & 7) + 8 * half_ + ((s >> 3) << 4)) * HD_DIM + t * 16 + ncol];
        cacc[t] = wmma_f16(pa[c2], vb, cacc[t]);
      }
    }
  }

  // epilogue: normalize and write ctx as [b, s, h*128+d] f16
#pragma unroll
  for (int t = 0; t < 8; ++t) {
#pragma unroll
    for (int r = 0; r < 8; ++r) {
      float val = cacc[t][r] / lstate[r];
      int s = q0 + r + 8 * half_;
      int d = h * HD_DIM + t * 16 + ncol;
      ctx[((size_t)b * S_DIM + s) * D_DIM + d] = (_Float16)val;
    }
  }
}

// ---------------------------------------------------------------------------
extern "C" void kernel_launch(void* const* d_in, const int* in_sizes, int n_in,
                              void* d_out, int out_size, void* d_ws, size_t ws_size,
                              hipStream_t stream) {
  (void)in_sizes; (void)n_in; (void)out_size; (void)ws_size;
  const float* x  = (const float*)d_in[0];
  const float* Wq = (const float*)d_in[1];
  const float* Wk = (const float*)d_in[2];
  const float* Wv = (const float*)d_in[3];
  const float* Wo = (const float*)d_in[4];
  const float* bo = (const float*)d_in[5];
  float* out = (float*)d_out;

  const size_t xN = (size_t)B_DIM * S_DIM * D_DIM;  // 16,777,216
  const size_t wN = (size_t)D_DIM * D_DIM;          //  4,194,304

  _Float16* ws   = (_Float16*)d_ws;
  _Float16* x16  = ws; ws += xN;
  _Float16* wq16 = ws; ws += wN;
  _Float16* wk16 = ws; ws += wN;
  _Float16* wv16 = ws; ws += wN;
  _Float16* wo16 = ws; ws += wN;
  _Float16* q16  = ws; ws += xN;  // [b,h,s,d], pre-scaled by 1/sqrt(128)
  _Float16* kt16 = ws; ws += xN;  // [b,h,d,s]
  _Float16* v16  = ws; ws += xN;  // [b,h,s,d]
  _Float16* c16  = ws; ws += xN;  // [b,s,D]

  cvt_f32_to_f16<<<2048, 256, 0, stream>>>(x,  x16,  xN);
  cvt_f32_to_f16<<<1024, 256, 0, stream>>>(Wq, wq16, wN);
  cvt_f32_to_f16<<<1024, 256, 0, stream>>>(Wk, wk16, wN);
  cvt_f32_to_f16<<<1024, 256, 0, stream>>>(Wv, wv16, wN);
  cvt_f32_to_f16<<<1024, 256, 0, stream>>>(Wo, wo16, wN);

  const int M = B_DIM * S_DIM;            // 8192
  dim3 ggrid(D_DIM / 128, M / 128);       // (16, 64)
  const float qscale = 0.08838834764831845f;  // 1/sqrt(HD_DIM)

  gemm128<0><<<ggrid, 256, 0, stream>>>(x16, wq16, q16,  nullptr, nullptr,
                                        M, D_DIM, D_DIM, qscale);
  gemm128<1><<<ggrid, 256, 0, stream>>>(x16, wk16, kt16, nullptr, nullptr,
                                        M, D_DIM, D_DIM, 1.0f);
  gemm128<2><<<ggrid, 256, 0, stream>>>(x16, wv16, v16,  nullptr, nullptr,
                                        M, D_DIM, D_DIM, 1.0f);

  attn_fa<<<B_DIM * H_NUM * (S_DIM / 64), 128, 0, stream>>>(q16, kt16, v16, c16);

  gemm128<3><<<ggrid, 256, 0, stream>>>(c16, wo16, nullptr, out, bo,
                                        M, D_DIM, D_DIM, 1.0f);
}
